// GloBEFFN_89593017795303
// MI455X (gfx1250) — matrix operand order repaired and checked by reading, not compiled, not run on hardware
//
#include <hip/hip_runtime.h>
#include <hip/hip_bf16.h>

typedef __attribute__((ext_vector_type(16))) __bf16 v16bf;
typedef __attribute__((ext_vector_type(8)))  float  v8f;

struct U128 { unsigned int a, b, c, d; };
union Frag { v16bf v; unsigned short s[16]; U128 q[2]; };

constexpr int S_ = 1024, D_ = 1024, P_ = 2816, R_ = 64, E_ = 8, NM_ = 16;

// workspace layout (bytes)
constexpr size_t OFF_AUP   = 0;
constexpr size_t OFF_AGT   = OFF_AUP + (size_t)E_*NM_*4;        // 512
constexpr size_t OFF_CNT   = OFF_AGT + (size_t)E_*NM_*4;        // 1024
constexpr size_t OFF_OFS   = OFF_CNT + (size_t)E_*4;            // 1056
constexpr size_t OFF_TLIST = 4096;
constexpr size_t OFF_TW    = OFF_TLIST + (size_t)E_*S_*4;
constexpr size_t OFF_HBF   = OFF_TW    + (size_t)E_*S_*4;
constexpr size_t OFF_UPMIX = OFF_HBF   + (size_t)S_*D_*2;
constexpr size_t OFF_GTMIX = OFF_UPMIX + (size_t)E_*D_*R_*2;
constexpr size_t OFF_UPAD  = OFF_GTMIX + (size_t)E_*D_*R_*2;
constexpr size_t OFF_GTAD  = OFF_UPAD  + (size_t)E_*P_*R_*2;
constexpr size_t OFF_INTER = OFF_GTAD  + (size_t)E_*P_*R_*2;    // ~15.8MB up to here
constexpr size_t OFF_DOWNBF= OFF_INTER + (size_t)S_*P_*2;
constexpr size_t WS_NEED_BF= OFF_DOWNBF + (size_t)E_*D_*P_*2;   // ~61.9MB total

__device__ inline unsigned short f2bf(float f) {
  unsigned int u = __float_as_uint(f);
  u += 0x7FFFu + ((u >> 16) & 1u);   // round-to-nearest-even
  return (unsigned short)(u >> 16);
}

__device__ inline v8f wmma_bf16(const Frag& a, const Frag& b, v8f c) {
  return __builtin_amdgcn_wmma_f32_16x16x32_bf16(false, a.v, false, b.v,
                                                 (short)0, c, false, false);
}

// ---- softmax of mixture logits + zero routing counters ----
__global__ void k_prep(const float* __restrict__ upl, const float* __restrict__ gtl,
                       float* aup, float* agt, int* counts) {
  int t = threadIdx.x;
  if (t < E_ * 2) {
    const float* l = (t < E_) ? upl : gtl;
    float* a       = (t < E_) ? aup : agt;
    int e          = (t < E_) ? t : t - E_;
    float mx = -1e30f;
    for (int m = 0; m < NM_; ++m) mx = fmaxf(mx, l[e*NM_+m]);
    float ex[NM_]; float sum = 0.f;
    for (int m = 0; m < NM_; ++m) { ex[m] = __expf(l[e*NM_+m] - mx); sum += ex[m]; }
    float inv = 1.f / sum;
    for (int m = 0; m < NM_; ++m) a[e*NM_+m] = ex[m] * inv;
  }
  if (t >= 32 && t < 32 + E_) counts[t - 32] = 0;
}

// ---- mixed bases: mixed[e,d,r] = sum_m alpha[e,m]*bank[m,d,r]; store bf16 TRANSPOSED [E,R,D] ----
__global__ void k_mixed(const float* __restrict__ aup, const float* __restrict__ agt,
                        const float* __restrict__ ubank, const float* __restrict__ gbank,
                        unsigned short* umix, unsigned short* gmix) {
  int gid = blockIdx.x * blockDim.x + threadIdx.x;
  if (gid >= E_*D_*R_) return;
  int e   = gid / (D_*R_);
  int rem = gid - e * (D_*R_);     // rem = d*R + r
  int d = rem / R_, r = rem - d*R_;
  float su = 0.f, sg = 0.f;
  #pragma unroll
  for (int m = 0; m < NM_; ++m) {
    su += aup[e*NM_+m] * ubank[(size_t)m*D_*R_ + rem];
    sg += agt[e*NM_+m] * gbank[(size_t)m*D_*R_ + rem];
  }
  size_t o = ((size_t)e*R_ + r) * D_ + d;   // transposed layout [E,R,D]
  umix[o] = f2bf(su); gmix[o] = f2bf(sg);
}

__global__ void k_h2bf(const float* __restrict__ h, unsigned short* o) {
  int g = blockIdx.x * blockDim.x + threadIdx.x;
  if (g < S_*D_) o[g] = f2bf(h[g]);
}

__global__ void k_ad2bf(const float* __restrict__ ua, const float* __restrict__ ga,
                        unsigned short* uo, unsigned short* go) {
  int g = blockIdx.x * blockDim.x + threadIdx.x;
  if (g < E_*P_*R_) { uo[g] = f2bf(ua[g]); go[g] = f2bf(ga[g]); }
}

// ---- one-time fp32->bf16 of down_projections (only if ws is big enough) ----
__global__ void k_down2bf(const float* __restrict__ dp, unsigned short* o) {
  long long g = (long long)blockIdx.x * blockDim.x + threadIdx.x;
  long long n = (long long)E_ * D_ * P_;
  if (g < n) o[g] = f2bf(dp[g]);
}

// ---- bucket tokens by expert (K=1) ----
__global__ void k_route(const int* __restrict__ idx, const float* __restrict__ wts,
                        int* counts, int* tlist, float* tw) {
  int s = blockIdx.x * blockDim.x + threadIdx.x;
  if (s >= S_) return;
  int e = idx[s] & (E_ - 1);
  int pos = atomicAdd(&counts[e], 1);
  tlist[e*S_ + pos] = s;
  tw[e*S_ + pos]    = wts[s];
}

__global__ void k_offsets(const int* __restrict__ counts, int* ofs) {
  if (threadIdx.x == 0 && blockIdx.x == 0) {
    int run = 0;
    for (int e = 0; e < E_; ++e) { ofs[e] = run; run += counts[e]; }
  }
}

// ---- stage 1+2 fused: T = Hg@mixed (D-reduction), then inter = silu(T_g@Ag^T)*(T_u@Au^T) ----
__global__ __launch_bounds__(32) void k_stage12(
    const unsigned short* __restrict__ hbf,
    const unsigned short* __restrict__ umix, const unsigned short* __restrict__ gmix,
    const unsigned short* __restrict__ uad,  const unsigned short* __restrict__ gad,
    const int* __restrict__ counts, const int* __restrict__ ofs,
    const int* __restrict__ tlist, unsigned short* inter) {
  int e = blockIdx.x >> 6;
  int tile = blockIdx.x & 63;
  int cnt = counts[e];
  if (tile * 16 >= cnt) return;                 // uniform exit, EXEC stays full
  int lane = threadIdx.x;
  int hf = lane >> 4;
  int m = lane & 15, n = lane & 15;
  int ti = tile*16 + m;
  int row = tlist[e*S_ + (ti < cnt ? ti : cnt - 1)];
  const unsigned short* hrow = hbf + (size_t)row * D_;
  const unsigned short* ub = umix + (size_t)e*R_*D_;   // [R,D]
  const unsigned short* gb = gmix + (size_t)e*R_*D_;

  v8f z = {0.f,0.f,0.f,0.f,0.f,0.f,0.f,0.f};
  v8f aU[4] = {z,z,z,z}, aG[4] = {z,z,z,z};

  // stage 1: [16 tok, 64 r] accumulators over D=1024
  for (int k0 = 0; k0 < D_; k0 += 32) {
    Frag A;
    int kb = k0 + (hf ? 16 : 0);
    A.q[0] = *(const U128*)(hrow + kb);
    A.q[1] = *(const U128*)(hrow + kb + 8);
    int ks = k0 + (hf ? 16 : 0);
    #pragma unroll
    for (int t = 0; t < 4; ++t) {
      int col = t*16 + n;
      Frag Bu, Bg;
      const unsigned short* uc = ub + (size_t)col * D_ + ks;
      const unsigned short* gc = gb + (size_t)col * D_ + ks;
      Bu.q[0] = *(const U128*)(uc);  Bu.q[1] = *(const U128*)(uc + 8);
      Bg.q[0] = *(const U128*)(gc);  Bg.q[1] = *(const U128*)(gc + 8);
      aU[t] = wmma_bf16(A, Bu, aU[t]);
      aG[t] = wmma_bf16(A, Bg, aG[t]);
    }
  }

  __shared__ unsigned short Tu[16*64];
  __shared__ unsigned short Tg[16*64];
  #pragma unroll
  for (int t = 0; t < 4; ++t)
    #pragma unroll
    for (int r = 0; r < 8; ++r) {
      int mm = (hf ? 8 : 0) + r;
      int col = t*16 + n;
      Tu[mm*64 + col] = f2bf(aU[t][r]);
      Tg[mm*64 + col] = f2bf(aG[t][r]);
    }
  __syncthreads();

  // stage 2: [16 tok, P] over K=R=64, chunks of 64 columns
  int cbase = ofs[e] + tile*16;
  for (int p0 = 0; p0 < P_; p0 += 64) {
    v8f oU[4] = {z,z,z,z}, oG[4] = {z,z,z,z};
    #pragma unroll
    for (int r0 = 0; r0 < R_; r0 += 32) {
      Frag Au, Ag;
      int kb = r0 + (hf ? 16 : 0);
      Au.q[0] = *(const U128*)&Tu[m*64 + kb];
      Au.q[1] = *(const U128*)&Tu[m*64 + kb + 8];
      Ag.q[0] = *(const U128*)&Tg[m*64 + kb];
      Ag.q[1] = *(const U128*)&Tg[m*64 + kb + 8];
      int rk = r0 + (hf ? 16 : 0);
      #pragma unroll
      for (int t = 0; t < 4; ++t) {
        int pcol = p0 + t*16 + n;
        Frag Bu, Bg;
        const unsigned short* up_ = uad + ((size_t)e*P_ + pcol) * R_ + rk;
        const unsigned short* gp_ = gad + ((size_t)e*P_ + pcol) * R_ + rk;
        Bu.q[0] = *(const U128*)(up_);  Bu.q[1] = *(const U128*)(up_ + 8);
        Bg.q[0] = *(const U128*)(gp_);  Bg.q[1] = *(const U128*)(gp_ + 8);
        oU[t] = wmma_bf16(Au, Bu, oU[t]);
        oG[t] = wmma_bf16(Ag, Bg, oG[t]);
      }
    }
    #pragma unroll
    for (int t = 0; t < 4; ++t)
      #pragma unroll
      for (int r = 0; r < 8; ++r) {
        int mm = (hf ? 8 : 0) + r;
        if (tile*16 + mm < cnt) {
          float up = oU[t][r], g = oG[t][r];
          float sg = g / (1.f + __expf(-g));        // silu
          inter[(size_t)(cbase + mm) * P_ + p0 + t*16 + n] = f2bf(sg * up);
        }
      }
  }
}

// ---- stage 3 (fallback): B converted fp32->bf16 in-register ----
__global__ __launch_bounds__(256) void k_down(
    const unsigned short* __restrict__ inter, const float* __restrict__ down,
    const int* __restrict__ counts, const int* __restrict__ ofs,
    const int* __restrict__ tlist, const float* __restrict__ tw,
    float* __restrict__ out) {
  int e = blockIdx.x >> 6;
  int tile = blockIdx.x & 63;
  int cnt = counts[e];
  if (tile * 16 >= cnt) return;                 // uniform exit
  int wave = threadIdx.x >> 5;
  int lane = threadIdx.x & 31;
  int hf = lane >> 4;
  int m = lane & 15, n = lane & 15;
  int cbase = ofs[e] + tile*16;
  int arow_i = cbase + m; if (arow_i > S_ - 1) arow_i = S_ - 1;
  const unsigned short* arow = inter + (size_t)arow_i * P_;
  const float* dmat = down + (size_t)e * D_ * P_;
  v8f z = {0.f,0.f,0.f,0.f,0.f,0.f,0.f,0.f};

  for (int dt = 0; dt < 8; ++dt) {
    int d0 = dt*128 + wave*16;                  // 8 waves x 8 tiles cover D=1024
    v8f acc = z;
    const float* brow = dmat + (size_t)(d0 + n) * P_ + (hf ? 16 : 0);
    for (int p0 = 0; p0 < P_; p0 += 32) {
      Frag A;
      int kb = p0 + (hf ? 16 : 0);
      A.q[0] = *(const U128*)(arow + kb);
      A.q[1] = *(const U128*)(arow + kb + 8);
      const float* bp = brow + p0;
      __builtin_prefetch(bp + 128, 0, 1);
      Frag B;
      #pragma unroll
      for (int k = 0; k < 16; ++k) B.s[k] = f2bf(bp[k]);   // fp32 -> bf16 in-register
      acc = wmma_bf16(A, B, acc);
    }
    #pragma unroll
    for (int r = 0; r < 8; ++r) {
      int mm = (hf ? 8 : 0) + r;
      int ti = tile*16 + mm;
      if (ti < cnt) {
        int trow = tlist[e*S_ + ti];
        out[(size_t)trow * D_ + d0 + n] = acc[r] * tw[e*S_ + ti];
      }
    }
  }
}

// ---- stage 3 (fast path): pre-converted bf16 down matrix, pure b128+wmma loop ----
__global__ __launch_bounds__(256) void k_down_bf(
    const unsigned short* __restrict__ inter, const unsigned short* __restrict__ dbf,
    const int* __restrict__ counts, const int* __restrict__ ofs,
    const int* __restrict__ tlist, const float* __restrict__ tw,
    float* __restrict__ out) {
  int e = blockIdx.x >> 6;
  int tile = blockIdx.x & 63;
  int cnt = counts[e];
  if (tile * 16 >= cnt) return;                 // uniform exit
  int wave = threadIdx.x >> 5;
  int lane = threadIdx.x & 31;
  int hf = lane >> 4;
  int m = lane & 15, n = lane & 15;
  int cbase = ofs[e] + tile*16;
  int arow_i = cbase + m; if (arow_i > S_ - 1) arow_i = S_ - 1;
  const unsigned short* arow = inter + (size_t)arow_i * P_;
  const unsigned short* dmat = dbf + (size_t)e * D_ * P_;
  v8f z = {0.f,0.f,0.f,0.f,0.f,0.f,0.f,0.f};

  for (int dt = 0; dt < 8; ++dt) {
    int d0 = dt*128 + wave*16;
    v8f acc = z;
    const unsigned short* brow = dmat + (size_t)(d0 + n) * P_ + (hf ? 16 : 0);
    for (int p0 = 0; p0 < P_; p0 += 32) {
      Frag A;
      int kb = p0 + (hf ? 16 : 0);
      A.q[0] = *(const U128*)(arow + kb);
      A.q[1] = *(const U128*)(arow + kb + 8);
      const unsigned short* bp = brow + p0;
      __builtin_prefetch(bp + 256, 0, 1);
      Frag B;
      B.q[0] = *(const U128*)(bp);
      B.q[1] = *(const U128*)(bp + 8);
      acc = wmma_bf16(A, B, acc);
    }
    #pragma unroll
    for (int r = 0; r < 8; ++r) {
      int mm = (hf ? 8 : 0) + r;
      int ti = tile*16 + mm;
      if (ti < cnt) {
        int trow = tlist[e*S_ + ti];
        out[(size_t)trow * D_ + d0 + n] = acc[r] * tw[e*S_ + ti];
      }
    }
  }
}

extern "C" void kernel_launch(void* const* d_in, const int* in_sizes, int n_in,
                              void* d_out, int out_size, void* d_ws, size_t ws_size,
                              hipStream_t stream) {
  (void)in_sizes; (void)n_in; (void)out_size;
  const float* h      = (const float*)d_in[0];
  const int*   eidx   = (const int*)  d_in[1];
  const float* ew     = (const float*)d_in[2];
  const float* upad_f = (const float*)d_in[3];
  const float* gtad_f = (const float*)d_in[4];
  const float* upl    = (const float*)d_in[5];
  const float* gtl    = (const float*)d_in[6];
  const float* downp  = (const float*)d_in[7];
  const float* ubank  = (const float*)d_in[8];
  const float* gbank  = (const float*)d_in[9];
  float* out = (float*)d_out;

  char* w = (char*)d_ws;
  float* aup   = (float*)(w + OFF_AUP);
  float* agt   = (float*)(w + OFF_AGT);
  int*   counts= (int*)  (w + OFF_CNT);
  int*   ofs   = (int*)  (w + OFF_OFS);
  int*   tlist = (int*)  (w + OFF_TLIST);
  float* tw    = (float*)(w + OFF_TW);
  unsigned short* hbf  = (unsigned short*)(w + OFF_HBF);
  unsigned short* umix = (unsigned short*)(w + OFF_UPMIX);
  unsigned short* gmix = (unsigned short*)(w + OFF_GTMIX);
  unsigned short* uad  = (unsigned short*)(w + OFF_UPAD);
  unsigned short* gad  = (unsigned short*)(w + OFF_GTAD);
  unsigned short* inter= (unsigned short*)(w + OFF_INTER);
  unsigned short* dbf  = (unsigned short*)(w + OFF_DOWNBF);

  const bool use_bf_down = (ws_size >= WS_NEED_BF);   // constant across calls -> deterministic

  k_prep  <<<1, 64, 0, stream>>>(upl, gtl, aup, agt, counts);
  k_mixed <<<(E_*D_*R_ + 255)/256, 256, 0, stream>>>(aup, agt, ubank, gbank, umix, gmix);
  k_h2bf  <<<(S_*D_ + 255)/256, 256, 0, stream>>>(h, hbf);
  k_ad2bf <<<(E_*P_*R_ + 255)/256, 256, 0, stream>>>(upad_f, gtad_f, uad, gad);
  if (use_bf_down) {
    long long nd = (long long)E_ * D_ * P_;
    k_down2bf<<<(unsigned int)((nd + 255) / 256), 256, 0, stream>>>(downp, dbf);
  }
  k_route <<<(S_ + 255)/256, 256, 0, stream>>>(eidx, ew, counts, tlist, tw);
  k_offsets<<<1, 1, 0, stream>>>(counts, ofs);
  k_stage12<<<E_*64, 32, 0, stream>>>(hbf, umix, gmix, uad, gad, counts, ofs, tlist, inter);
  if (use_bf_down)
    k_down_bf<<<E_*64, 256, 0, stream>>>(inter, dbf, counts, ofs, tlist, tw, out);
  else
    k_down  <<<E_*64, 256, 0, stream>>>(inter, downp, counts, ofs, tlist, tw, out);
}
